// MyTestNetwork_64673617543220
// MI455X (gfx1250) — matrix operand order repaired and checked by reading, not compiled, not run on hardware
//
#include <hip/hip_runtime.h>
#include <hip/hip_bf16.h>

// ---------------------------------------------------------------------------
// Continuous convolution (Open3D CConv style) for MI455X / gfx1250.
//   msg[e] = sum_corners w_corner * (feat[e] @ W2[idx_corner])   (3 -> 32 ch)
//   out    = segment_sum(msg, dst) + b
// Strategy: per-wave batches of 32 edges; expand each edge to a 192-wide
// sparse row A[e][cell*3+c] = w_cell*feat[c] in LDS (bf16) and run a
// (32 x 192) @ (192 x 32) GEMM with v_wmma_f32_16x16x32_bf16 (f32 accum).
// dst is sorted -> run-length-compressed atomic scatter.
// ---------------------------------------------------------------------------

typedef __attribute__((ext_vector_type(16))) __bf16 v16bf;
typedef __attribute__((ext_vector_type(8)))  float  v8f;

#define WAVES 4
#define BLOCK (WAVES * 32)

#define C_OUT 32
#define KDIM  192            // 64 kernel cells * 3 input channels
#define A_ROW_BYTES (KDIM * 2)          // 384 B per edge row (bf16)
#define WB_BYTES  (6 * 2 * 32 * 32)     // 6 k-steps * 2 n-tiles * 32 lanes * 32B
#define A_BYTES   (WAVES * 32 * A_ROW_BYTES)
#define DST_BYTES (WAVES * 32 * 4)

__device__ __forceinline__ float signf(float v) {
  return (v > 0.f) ? 1.f : ((v < 0.f) ? -1.f : 0.f);
}

__global__ void cconv_init_out(float* __restrict__ out,
                               const float* __restrict__ b, int n) {
  int i = blockIdx.x * blockDim.x + threadIdx.x;
  if (i < n) out[i] = b[i & (C_OUT - 1)];
}

__global__ __launch_bounds__(BLOCK)
void cconv_edge_kernel(const float* __restrict__ vel,
                       const float* __restrict__ pos0,
                       const float* __restrict__ pos1,
                       const float* __restrict__ W,      // (4,4,4,3,32) = (192,32)
                       const int*   __restrict__ src,
                       const int*   __restrict__ dst,
                       const float* __restrict__ mask,
                       float* __restrict__ out,          // (8192,32), pre-set to bias
                       int E) {
  constexpr float INV_RADIUS   = 1.0f / (1.5f * 6.0f * 0.025f * 0.5f); // 1/0.1125
  constexpr float EPS          = 1e-12f;
  constexpr float FOUR_OVER_PI = 1.2732395447351628f;

  __shared__ __align__(16) unsigned char smem[WB_BYTES + A_BYTES + DST_BYTES];
  __bf16*        Wb       = (__bf16*)smem;
  unsigned char* AbaseAll = smem + WB_BYTES;
  int*           dstS     = (int*)(smem + WB_BYTES + A_BYTES);

  const int tid   = threadIdx.x;
  const int lane  = tid & 31;
  const int wib   = tid >> 5;      // wave index in block
  const int lhalf = lane >> 4;     // 0 | 1 (half-wave)
  const int l16   = lane & 15;

  // ---- stage W (192x32 f32) into bf16 B-fragment order, once per block ----
  // layout: Wb[((ks*2 + nt)*32 + lane)*16 + j] = W[(32*ks + 8*(lane>>4) + (j&7) + 16*(j>>3))*32
  //                                               + (lane&15) + 16*nt]
  for (int t = tid; t < 6 * 2 * 32 * 16; t += BLOCK) {
    int j  = t & 15;
    int ln = (t >> 4) & 31;
    int nt = (t >> 9) & 1;
    int ks = t >> 10;
    int K  = 32 * ks + 8 * (ln >> 4) + (j & 7) + 16 * (j >> 3);
    int n  = (ln & 15) + 16 * nt;
    Wb[t] = (__bf16)W[K * C_OUT + n];
  }
  __syncthreads();

  uint4*       Arow_u4 = (uint4*)(AbaseAll + wib * 32 * A_ROW_BYTES); // 24 uint4/row
  const uint4* Wb_u4   = (const uint4*)Wb;

  const int nBatch = (E + 31) >> 5;
  const int gwarp  = (int)((blockIdx.x * BLOCK + tid) >> 5);
  const int nWarp  = (int)((gridDim.x * BLOCK) >> 5);

  for (int bt = gwarp; bt < nBatch; bt += nWarp) {
    const int e = bt * 32 + lane;

    // ------------------- per-edge geometry (one edge per lane) -------------
    int   si = 0, di = 0;
    float m  = 0.f;
    if (e < E) { si = src[e]; di = dst[e]; m = mask[e]; }

    float x = (pos0[si * 3 + 0] - pos1[di * 3 + 0]) * INV_RADIUS;
    float y = (pos0[si * 3 + 1] - pos1[di * 3 + 1]) * INV_RADIUS;
    float z = (pos0[si * 3 + 2] - pos1[di * 3 + 2]) * INV_RADIUS;

    float sq  = x * x + y * y + z * z;
    float t1  = 1.0f - sq;
    float win = (sq < 1.0f) ? (t1 * t1 * t1) * m : 0.0f;

    float norm = sqrtf(sq + EPS);
    float xy2  = x * x + y * y;
    bool  top  = (1.25f * z * z) > xy2;
    float sTop = sqrtf(3.0f * norm / (norm + fabsf(z) + EPS));
    float sSid = norm / sqrtf(xy2 + EPS);
    float cx   = top ? x * sTop : x * sSid;
    float cy   = top ? y * sTop : y * sSid;
    float cz   = top ? signf(z) * norm : 1.5f * z;
    if (sq < 1e-12f) { cx = 0.f; cy = 0.f; cz = 0.f; }

    float nxy = sqrtf(cx * cx + cy * cy + EPS);
    float xs  = (fabsf(cx) > EPS) ? cx : 1.0f;
    float ys  = (fabsf(cy) > EPS) ? cy : 1.0f;
    bool  c1  = (cx * cx) >= (cy * cy);
    float num = c1 ? cy : cx;
    float den = c1 ? xs : ys;
    float at  = FOUR_OVER_PI * atanf(num / den);
    float sg  = c1 ? signf(cx) : signf(cy);
    float ux  = c1 ? sg * nxy : sg * nxy * at;
    float uy  = c1 ? sg * nxy * at : sg * nxy;
    if (cx * cx + cy * cy < 1e-12f) { ux = 0.f; uy = 0.f; }

    float gx = fminf(fmaxf((ux + 1.0f) * 1.5f, 0.0f), 3.0f);
    float gy = fminf(fmaxf((uy + 1.0f) * 1.5f, 0.0f), 3.0f);
    float gz = fminf(fmaxf((cz + 1.0f) * 1.5f, 0.0f), 3.0f);
    int   ix = (int)fminf(floorf(gx), 2.0f);
    int   iy = (int)fminf(floorf(gy), 2.0f);
    int   iz = (int)fminf(floorf(gz), 2.0f);
    float fx = gx - (float)ix;
    float fy = gy - (float)iy;
    float fz = gz - (float)iz;

    float v0 = vel[si * 3 + 0];
    float v1 = vel[si * 3 + 1];
    float v2 = vel[si * 3 + 2];

    // ------------------- build sparse A row in LDS (bf16) ------------------
    uint4* Ar = Arow_u4 + lane * 24;
    uint4  zz; zz.x = 0u; zz.y = 0u; zz.z = 0u; zz.w = 0u;
#pragma unroll
    for (int i = 0; i < 24; ++i) Ar[i] = zz;   // clear 192 bf16 (ds_store_b128)

    __bf16* ArB = (__bf16*)Ar;
#pragma unroll
    for (int dz2 = 0; dz2 < 2; ++dz2) {
      float wz = dz2 ? fz : (1.0f - fz);
#pragma unroll
      for (int dy2 = 0; dy2 < 2; ++dy2) {
        float wy = dy2 ? fy : (1.0f - fy);
#pragma unroll
        for (int dx2 = 0; dx2 < 2; ++dx2) {
          float wx = dx2 ? fx : (1.0f - fx);
          float w  = wz * wy * wx * win;
          int cell = ((iz + dz2) * 4 + (iy + dy2)) * 4 + (ix + dx2);
          int c3   = cell * 3;
          ArB[c3 + 0] = (__bf16)(w * v0);
          ArB[c3 + 1] = (__bf16)(w * v1);
          ArB[c3 + 2] = (__bf16)(w * v2);
        }
      }
    }
    dstS[wib * 32 + lane] = di;
    __builtin_amdgcn_wave_barrier();   // keep frag loads after the LDS scatter

    // ------------------- WMMA: (32 x 192) @ (192 x 32), f32 accum ----------
    v8f acc00 = {}, acc01 = {}, acc10 = {}, acc11 = {};
#pragma unroll
    for (int ks = 0; ks < 6; ++ks) {
      union { uint4 u[2]; v16bf v; } a0, a1, b0, b1;
      // A frag: row r, K-chunks at 8*(lane/16) and 16+8*(lane/16) within k-step
      int colu = 4 * ks + lhalf;                       // uint4 index in row
      a0.u[0] = Arow_u4[(0  + l16) * 24 + colu];
      a0.u[1] = Arow_u4[(0  + l16) * 24 + colu + 2];
      a1.u[0] = Arow_u4[(16 + l16) * 24 + colu];
      a1.u[1] = Arow_u4[(16 + l16) * 24 + colu + 2];
      // B frag: 32 contiguous bytes per lane, prestaged
      b0.u[0] = Wb_u4[((ks * 2 + 0) * 32 + lane) * 2 + 0];
      b0.u[1] = Wb_u4[((ks * 2 + 0) * 32 + lane) * 2 + 1];
      b1.u[0] = Wb_u4[((ks * 2 + 1) * 32 + lane) * 2 + 0];
      b1.u[1] = Wb_u4[((ks * 2 + 1) * 32 + lane) * 2 + 1];

      acc00 = __builtin_amdgcn_wmma_f32_16x16x32_bf16(false, a0.v, false, b0.v,
                                                      (short)0, acc00, false, false);
      acc01 = __builtin_amdgcn_wmma_f32_16x16x32_bf16(false, a0.v, false, b1.v,
                                                      (short)0, acc01, false, false);
      acc10 = __builtin_amdgcn_wmma_f32_16x16x32_bf16(false, a1.v, false, b0.v,
                                                      (short)0, acc10, false, false);
      acc11 = __builtin_amdgcn_wmma_f32_16x16x32_bf16(false, a1.v, false, b1.v,
                                                      (short)0, acc11, false, false);
    }
    __builtin_amdgcn_wave_barrier();

    // ------------------- run-length-compressed atomic scatter --------------
    // lane owns channel l16 (+0/+16) for 8 consecutive edge rows per tile;
    // dst is sorted, so merge equal-dst runs before hitting L2 atomics.
    const int rbase = wib * 32;
#pragma unroll
    for (int mt = 0; mt < 2; ++mt) {
      v8f aN0 = mt ? acc10 : acc00;
      v8f aN1 = mt ? acc11 : acc01;
      int r0  = mt * 16 + lhalf * 8;
      int dprev = dstS[rbase + r0];
      float s0 = aN0[0];
      float s1 = aN1[0];
#pragma unroll
      for (int v = 1; v < 8; ++v) {
        int dn = dstS[rbase + r0 + v];
        if (dn == dprev) {
          s0 += aN0[v];
          s1 += aN1[v];
        } else {
          __hip_atomic_fetch_add(&out[dprev * C_OUT + l16], s0,
                                 __ATOMIC_RELAXED, __HIP_MEMORY_SCOPE_AGENT);
          __hip_atomic_fetch_add(&out[dprev * C_OUT + 16 + l16], s1,
                                 __ATOMIC_RELAXED, __HIP_MEMORY_SCOPE_AGENT);
          dprev = dn; s0 = aN0[v]; s1 = aN1[v];
        }
      }
      __hip_atomic_fetch_add(&out[dprev * C_OUT + l16], s0,
                             __ATOMIC_RELAXED, __HIP_MEMORY_SCOPE_AGENT);
      __hip_atomic_fetch_add(&out[dprev * C_OUT + 16 + l16], s1,
                             __ATOMIC_RELAXED, __HIP_MEMORY_SCOPE_AGENT);
    }
  }
}

extern "C" void kernel_launch(void* const* d_in, const int* in_sizes, int n_in,
                              void* d_out, int out_size, void* d_ws, size_t ws_size,
                              hipStream_t stream) {
  const float* vel  = (const float*)d_in[0];
  const float* pos0 = (const float*)d_in[1];
  const float* pos1 = (const float*)d_in[2];
  const float* W    = (const float*)d_in[3];
  const float* b    = (const float*)d_in[4];
  const int*   src  = (const int*)d_in[5];
  const int*   dst  = (const int*)d_in[6];
  const float* mask = (const float*)d_in[7];
  float* out = (float*)d_out;
  const int E = in_sizes[5];

  cconv_init_out<<<(out_size + 255) / 256, 256, 0, stream>>>(out, b, out_size);

  const int blocks = 1024;   // 4096 waves, ~4.6 batches of 32 edges each
  cconv_edge_kernel<<<blocks, BLOCK, 0, stream>>>(vel, pos0, pos1, W, src, dst,
                                                  mask, out, E);
}
